// FactorNet_69518340653114
// MI455X (gfx1250) — matrix operand order repaired and checked by reading, not compiled, not run on hardware
//
#include <hip/hip_runtime.h>
#include <hip/hip_bf16.h>
#include <math.h>

// ---------------------------------------------------------------------------
// FactorNet fused kernel for MI455X (gfx1250), bf16 WMMA w/ f32 accumulate.
// 256 threads = 8 wave32, 128 rows/block, double-buffered per-d weight staging.
// ---------------------------------------------------------------------------

typedef __bf16 bf16;
typedef bf16  v16bf __attribute__((ext_vector_type(16)));
typedef float v8f   __attribute__((ext_vector_type(8)));

#define NROWS 16384
#define DD    64
#define H1D   64
#define H2D   64
#define OO    32
#define MIN_SIGMA 0.008f      // 0.001 * sqrt(64)
#define RPB   128             // rows per block
#define NTHR  256             // 8 wave32

// Packed-weight sizes (bf16 elements)
#define W2B_ELEMS  (DD * 8 * 32 * 16)   // 262144  (4 ntiles x 2 ksteps per d)
#define WMB_ELEMS  (DD * 4 * 32 * 16)   // 131072  (2 ntiles x 2 ksteps per d)

// ---------------------------------------------------------------------------
// One-time pack: f32 weights -> bf16 in exact WMMA B-fragment per-lane order.
// B (32x16, 16-bit): lanes 0-15 hold col=lane, K=0..7,16..23 (2 per VGPR);
//                    lanes 16-31 hold col=lane-16, K=8..15,24..31.
// Layout: [d][tile = nt*2+kt][lane 0..31][elem 0..15]
// ---------------------------------------------------------------------------
__global__ void pack_weights_kernel(const float* __restrict__ W2,
                                    const float* __restrict__ Wmu,
                                    const float* __restrict__ Wsig,
                                    bf16* __restrict__ W2B,
                                    bf16* __restrict__ WmuB,
                                    bf16* __restrict__ WsigB) {
  int idx = blockIdx.x * blockDim.x + threadIdx.x;
  if (idx < W2B_ELEMS) {
    int e  = idx & 15;
    int l  = (idx >> 4) & 31;
    int t  = (idx >> 9) & 7;
    int d  = idx >> 12;
    int nt = t >> 1, kt = t & 1;
    int n  = nt * 16 + (l & 15);
    int kk = kt * 32 + (e >> 3) * 16 + (l >> 4) * 8 + (e & 7);   // K index (== h1 index)
    W2B[idx] = (bf16)W2[(d * H1D + kk) * H2D + n];
  }
  if (idx < WMB_ELEMS) {
    int e  = idx & 15;
    int l  = (idx >> 4) & 31;
    int t  = (idx >> 9) & 3;
    int d  = idx >> 11;
    int nt = t >> 1, kt = t & 1;
    int n  = nt * 16 + (l & 15);
    int kk = kt * 32 + (e >> 3) * 16 + (l >> 4) * 8 + (e & 7);   // K index (== h2 index)
    WmuB[idx]  = (bf16)Wmu[(d * H2D + kk) * OO + n];
    WsigB[idx] = (bf16)Wsig[(d * H2D + kk) * OO + n];
  }
}

// softplus via native v_exp_f32 / v_log_f32 (log2-based HW transcendentals)
__device__ __forceinline__ float softplus_fast(float v) {
  const float LOG2E = 1.4426950408889634f;
  const float LN2   = 0.6931471805599453f;
  float t = __builtin_amdgcn_exp2f(-fabsf(v) * LOG2E);     // e^{-|v|}
  return fmaxf(v, 0.f) + LN2 * __builtin_amdgcn_logf(1.f + t);
}

// ---------------------------------------------------------------------------
// Main fused kernel.
// ---------------------------------------------------------------------------
__global__ __launch_bounds__(NTHR)
void factor_net_kernel(const float* __restrict__ x,
                       const float* __restrict__ W1,
                       const float* __restrict__ b1,
                       const float* __restrict__ b2,
                       const float* __restrict__ bmu,
                       const float* __restrict__ bsig,
                       const int*   __restrict__ mask,
                       const bf16*  __restrict__ W2B,
                       const bf16*  __restrict__ WmuB,
                       const bf16*  __restrict__ WsigB,
                       float* __restrict__ out) {
  __shared__ __align__(16) float xs[RPB][68];          // conflict-free column reads
  __shared__ __align__(32) bf16  w2s [2][8 * 512];     // double-buffered B-tiles
  __shared__ __align__(32) bf16  wmus[2][4 * 512];
  __shared__ __align__(32) bf16  wsgs[2][4 * 512];
  __shared__ __align__(16) bf16  h2s[8][16][80];       // per-wave h2 strip
  __shared__ float w1s[2][64], b1s[2][64], b2s[2][64];
  __shared__ float msk[2][RPB];
  __shared__ float bmus[2][32], bsigs[2][32];

  const int tid  = threadIdx.x;
  const int wave = tid >> 5;        // 0..7
  const int lane = tid & 31;
  const int hrow = lane & 15;       // A: row; B/C/D: column
  const int half = lane >> 4;       // lane half selects K / M offset
  const int kh   = half * 8;        // A-fragment K sub-offset
  const int row_base = blockIdx.x * RPB;

  // ---- stage x tile: 128 rows x 64 cols, coalesced float4 ----
  {
    const float4* xg = (const float4*)(x + row_base * DD);
#pragma unroll
    for (int it = 0; it < 8; ++it) {
      int i  = tid + it * NTHR;      // 0..2047 float4s
      int r  = i >> 4;
      int c4 = i & 15;
      float4 v = xg[i];
      *(float4*)&xs[r][c4 * 4] = v;
    }
  }

  // ---- per-d staging helpers (registers <-> LDS, double buffered) ----
  uint4 r2a, r2b, rmu, rsg;
  float sc_m = 0.f, sc_o = 0.f;

  auto stage_regs = [&](int d) {
    const uint4* s2 = (const uint4*)(W2B + d * 4096);   // 512 uint4
    r2a = s2[tid];
    r2b = s2[tid + NTHR];
    rmu = ((const uint4*)(WmuB  + d * 2048))[tid];      // 256 uint4
    rsg = ((const uint4*)(WsigB + d * 2048))[tid];
    if (tid < RPB) sc_m = (mask[(row_base + tid) * DD + d] != 0) ? 1.f : 0.f;
    if      (tid <  64) sc_o = W1  [d * H1D + tid];
    else if (tid < 128) sc_o = b1  [d * H1D + (tid - 64)];
    else if (tid < 192) sc_o = b2  [d * H2D + (tid - 128)];
    else if (tid < 224) sc_o = bmu [d * OO  + (tid - 192)];
    else                sc_o = bsig[d * OO  + (tid - 224)];
    if (d + 1 < DD) {            // keep next-next-d weights moving toward L2/L0
      __builtin_prefetch(W2B  + (d + 1) * 4096, 0, 1);
      __builtin_prefetch(WmuB + (d + 1) * 2048, 0, 1);
      __builtin_prefetch(WsigB + (d + 1) * 2048, 0, 1);
    }
  };
  auto store_stage = [&](int buf) {
    ((uint4*)w2s[buf])[tid]        = r2a;
    ((uint4*)w2s[buf])[tid + NTHR] = r2b;
    ((uint4*)wmus[buf])[tid]       = rmu;
    ((uint4*)wsgs[buf])[tid]       = rsg;
    if (tid < RPB) msk[buf][tid] = sc_m;
    if      (tid <  64) w1s  [buf][tid]       = sc_o;
    else if (tid < 128) b1s  [buf][tid - 64]  = sc_o;
    else if (tid < 192) b2s  [buf][tid - 128] = sc_o;
    else if (tid < 224) bmus [buf][tid - 192] = sc_o;
    else                bsigs[buf][tid - 224] = sc_o;
  };

  float np1[2][8], np2[2][8];
#pragma unroll
  for (int nt = 0; nt < 2; ++nt)
#pragma unroll
    for (int j = 0; j < 8; ++j) { np1[nt][j] = 0.f; np2[nt][j] = 0.f; }

  // prologue: fill buffer 0
  stage_regs(0);
  store_stage(0);
  __syncthreads();

  for (int d = 0; d < DD; ++d) {
    const int buf = d & 1;
    if (d + 1 < DD) stage_regs(d + 1);   // global loads in flight under compute

    // ---- h1 A-fragments built in registers: relu(x*W1+b1) -> bf16 ----
    const float xval = xs[wave * 16 + hrow][d];
    v16bf a10, a11;
#pragma unroll
    for (int j = 0; j < 8; ++j) {
      a10[j]     = (bf16)fmaxf(xval * w1s[buf][     kh + j] + b1s[buf][     kh + j], 0.f);
      a10[8 + j] = (bf16)fmaxf(xval * w1s[buf][16 + kh + j] + b1s[buf][16 + kh + j], 0.f);
      a11[j]     = (bf16)fmaxf(xval * w1s[buf][32 + kh + j] + b1s[buf][32 + kh + j], 0.f);
      a11[8 + j] = (bf16)fmaxf(xval * w1s[buf][48 + kh + j] + b1s[buf][48 + kh + j], 0.f);
    }

    // ---- GEMM1: h2_pre(16x64) = h1(16x64) @ W2[d](64x64); relu -> LDS bf16 ----
#pragma unroll
    for (int nt = 0; nt < 4; ++nt) {
      v8f c = {};
      const v16bf bA = *(const v16bf*)(&w2s[buf][((nt * 2 + 0) * 32 + lane) * 16]);
      const v16bf bB = *(const v16bf*)(&w2s[buf][((nt * 2 + 1) * 32 + lane) * 16]);
      c = __builtin_amdgcn_wmma_f32_16x16x32_bf16(false, a10, false, bA, (short)0, c, false, false);
      c = __builtin_amdgcn_wmma_f32_16x16x32_bf16(false, a11, false, bB, (short)0, c, false, false);
      const int col = nt * 16 + hrow;                // C/D: lane holds column
      const float b2v = b2s[buf][col];
#pragma unroll
      for (int j = 0; j < 8; ++j)                    // VGPR j holds row j (+8 upper half)
        h2s[wave][j + half * 8][col] = (bf16)fmaxf(c[j] + b2v, 0.f);
    }
    __builtin_amdgcn_wave_barrier();                 // per-wave strip; DS in-order per wave

    // ---- regather h2 as A-fragments ----
    v16bf a20, a21;
#pragma unroll
    for (int j = 0; j < 8; ++j) {
      a20[j]     = h2s[wave][hrow][     kh + j];
      a20[8 + j] = h2s[wave][hrow][16 + kh + j];
      a21[j]     = h2s[wave][hrow][32 + kh + j];
      a21[8 + j] = h2s[wave][hrow][48 + kh + j];
    }

    // ---- GEMM2/3: mu_pre, sig_pre (16x32) = h2 @ Wmu/Wsig[d](64x32) ----
    v8f m[2], sg[2];
#pragma unroll
    for (int nt = 0; nt < 2; ++nt) {
      v8f cm = {}, cs = {};
      const v16bf bm0 = *(const v16bf*)(&wmus[buf][((nt * 2 + 0) * 32 + lane) * 16]);
      const v16bf bm1 = *(const v16bf*)(&wmus[buf][((nt * 2 + 1) * 32 + lane) * 16]);
      const v16bf bs0 = *(const v16bf*)(&wsgs[buf][((nt * 2 + 0) * 32 + lane) * 16]);
      const v16bf bs1 = *(const v16bf*)(&wsgs[buf][((nt * 2 + 1) * 32 + lane) * 16]);
      cm = __builtin_amdgcn_wmma_f32_16x16x32_bf16(false, a20, false, bm0, (short)0, cm, false, false);
      cm = __builtin_amdgcn_wmma_f32_16x16x32_bf16(false, a21, false, bm1, (short)0, cm, false, false);
      cs = __builtin_amdgcn_wmma_f32_16x16x32_bf16(false, a20, false, bs0, (short)0, cs, false, false);
      cs = __builtin_amdgcn_wmma_f32_16x16x32_bf16(false, a21, false, bs1, (short)0, cs, false, false);
      m[nt] = cm; sg[nt] = cs;
    }

    // ---- f32 epilogue: softplus sigma, masked natural-parameter accumulation ----
#pragma unroll
    for (int nt = 0; nt < 2; ++nt) {
      const int col = nt * 16 + hrow;                // output index o
      const float bm = bmus[buf][col], bs = bsigs[buf][col];
#pragma unroll
      for (int j = 0; j < 8; ++j) {
        const int r = j + half * 8;
        const float mv  = msk[buf][wave * 16 + r];
        const float mu  = m[nt][j] + bm;
        const float sig = MIN_SIGMA + softplus_fast(sg[nt][j] + bs);
        const float rs  = __builtin_amdgcn_rcpf(sig);      // v_rcp_f32
        const float iv  = mv * rs * rs;                    // masked 1/sigma^2
        np1[nt][j] += mu * iv;
        np2[nt][j] -= 0.5f * iv;
      }
    }

    __syncthreads();                      // all waves done reading lds[buf]
    if (d + 1 < DD) store_stage(buf ^ 1); // publish next-d weights
    __syncthreads();                      // visible before next compute
  }

  // ---- finalize: sigma_out = sqrt(-0.5/np2); mu_out = np1 * sigma_out^2 ----
  float* out_mu  = out;
  float* out_sig = out + NROWS * OO;
#pragma unroll
  for (int nt = 0; nt < 2; ++nt) {
    const int col = nt * 16 + hrow;
#pragma unroll
    for (int j = 0; j < 8; ++j) {
      const int r    = j + half * 8;
      const int grow = row_base + wave * 16 + r;
      const float s2 = -0.5f * __builtin_amdgcn_rcpf(np2[nt][j]);
      out_mu [grow * OO + col] = np1[nt][j] * s2;
      out_sig[grow * OO + col] = __builtin_amdgcn_sqrtf(s2);
    }
  }
}

// ---------------------------------------------------------------------------
extern "C" void kernel_launch(void* const* d_in, const int* in_sizes, int n_in,
                              void* d_out, int out_size, void* d_ws, size_t ws_size,
                              hipStream_t stream) {
  (void)in_sizes; (void)n_in; (void)out_size; (void)ws_size;
  const float* x    = (const float*)d_in[0];
  const float* W1   = (const float*)d_in[1];
  const float* b1   = (const float*)d_in[2];
  const float* W2   = (const float*)d_in[3];
  const float* b2   = (const float*)d_in[4];
  const float* Wmu  = (const float*)d_in[5];
  const float* bmu  = (const float*)d_in[6];
  const float* Wsig = (const float*)d_in[7];
  const float* bsig = (const float*)d_in[8];
  const int*   mask = (const int*)d_in[9];

  bf16* W2B   = (bf16*)d_ws;            // 512 KB
  bf16* WmuB  = W2B + W2B_ELEMS;        // 256 KB
  bf16* WsigB = WmuB + WMB_ELEMS;       // 256 KB

  pack_weights_kernel<<<(W2B_ELEMS + 255) / 256, 256, 0, stream>>>(
      W2, Wmu, Wsig, W2B, WmuB, WsigB);
  factor_net_kernel<<<NROWS / RPB, NTHR, 0, stream>>>(
      x, W1, b1, b2, bmu, bsig, mask, W2B, WmuB, WsigB, (float*)d_out);
}
